// Derivative_20323785245150
// MI455X (gfx1250) — matrix operand order repaired and checked by reading, not compiled, not run on hardware
//
#include <hip/hip_runtime.h>
#include <hip/hip_bf16.h>

#define Bsz 4096
#define Dd  128
#define Hh  256
#define NSPLIT 4                 // batch splits per net
#define NTILES (Bsz / 128 / NSPLIT)  // 8 tiles of 128 rows per block

typedef __attribute__((ext_vector_type(8)))  __bf16 v8bf;
typedef __attribute__((ext_vector_type(16))) __bf16 v16bf;
typedef __attribute__((ext_vector_type(8)))  float  v8f;

// LDS row strides (bf16 elements), padded to dodge bank conflicts
#define XA_S 136   // 128 + 8
#define W1_S 136
#define W2_S 264   // 256 + 8
#define H1_S 264

__device__ __forceinline__ __bf16 f2bf(float x) { return (__bf16)x; }

__device__ __forceinline__ v16bf ld_frag(const __bf16* p0, const __bf16* p1) {
  v8bf a = *(const v8bf*)p0;   // ds_load_b128
  v8bf b = *(const v8bf*)p1;   // ds_load_b128
  return __builtin_shufflevector(a, b, 0,1,2,3,4,5,6,7,8,9,10,11,12,13,14,15);
}

__global__ __launch_bounds__(256) void mlp_ensemble_kernel(
    const float* __restrict__ inputs, const float* __restrict__ adj,
    const float* __restrict__ W1, const float* __restrict__ b1,
    const float* __restrict__ W2, const float* __restrict__ b2,
    const float* __restrict__ W3, const float* __restrict__ b3,
    float* __restrict__ out)
{
  extern __shared__ __align__(16) char smem[];
  __bf16* XA   = (__bf16*)smem;              // 128 x 136 bf16 : input batch tile
  __bf16* W1t  = XA  + Dd * XA_S;            // 256 x 136 bf16 : masked W1 transposed [h][d]
  __bf16* W2t  = W1t + Hh * W1_S;            // 256 x 264 bf16 : W2 transposed [k][h]
  __bf16* H1   = W2t + Hh * W2_S;            // 128 x 264 bf16 : layer-1 activations
  float*  b1s  = (float*)(H1 + Dd * H1_S);   // 256 f32
  float*  b2s  = b1s + Hh;                   // 256 f32
  float*  w3s  = b2s + Hh;                   // 256 f32
  float*  outp = w3s + Hh;                   // 128 f32 partial outputs

  const int i    = blockIdx.x;        // net index (0..127)
  const int t    = threadIdx.x;
  const int lane = t & 31;
  const int wave = t >> 5;
  const int hi   = lane >> 4;         // half-wave
  const int lr   = lane & 15;
  const int mg   = wave >> 2;         // 0..1 : row group (64 rows)
  const int ng   = wave & 3;          // 0..3 : col group (64 cols)

  // ---------------- stage weights once per block ----------------
  {
    b1s[t] = b1[i * Hh + t];
    b2s[t] = b2[i * Hh + t];
    w3s[t] = W3[i * Hh + t];

    const float* W1i  = W1  + (size_t)i * Dd * Hh;
    const float* adji = adj + (size_t)i * Dd;
    #pragma unroll 4
    for (int idx = t; idx < Dd * Hh; idx += 256) {
      int d = idx >> 8, h = idx & 255;
      W1t[h * W1_S + d] = f2bf(W1i[d * Hh + h] * adji[d]);  // fold adjacency mask into W1
    }
    const float* W2i = W2 + (size_t)i * Hh * Hh;
    #pragma unroll 4
    for (int idx = t; idx < Hh * Hh; idx += 256) {
      int h = idx >> 8, k = idx & 255;
      W2t[k * W2_S + h] = f2bf(W2i[h * Hh + k]);
    }
  }
  const float b3i = b3[i];

  for (int tl = 0; tl < NTILES; ++tl) {
    const int b0 = (blockIdx.y * NTILES + tl) * 128;

    // ---------------- stage input tile (8 bf16 per ds_store_b128) ----------------
    if (t < 128) outp[t] = 0.0f;
    {
      const float* inp = inputs + (size_t)b0 * Dd;
      #pragma unroll
      for (int g = t; g < (Dd * Dd) / 8; g += 256) {   // 8 iterations
        int r = g >> 4, d8 = (g & 15) * 8;
        const float4 u0 = *(const float4*)(inp + r * Dd + d8);
        const float4 u1 = *(const float4*)(inp + r * Dd + d8 + 4);
        v8bf v;
        v[0] = f2bf(u0.x); v[1] = f2bf(u0.y); v[2] = f2bf(u0.z); v[3] = f2bf(u0.w);
        v[4] = f2bf(u1.x); v[5] = f2bf(u1.y); v[6] = f2bf(u1.z); v[7] = f2bf(u1.w);
        *(v8bf*)(XA + r * XA_S + d8) = v;
      }
    }
    __syncthreads();

    // ---------------- layer 1: [128x128] @ [128x256] ----------------
    v8f acc[4][4] = {};
    #pragma unroll
    for (int ks = 0; ks < 4; ++ks) {
      const int k0 = ks * 32;
      v16bf bf[4];
      #pragma unroll
      for (int nt = 0; nt < 4; ++nt) {
        const __bf16* bp = W1t + (ng * 64 + nt * 16 + lr) * W1_S + k0 + hi * 8;
        bf[nt] = ld_frag(bp, bp + 16);
      }
      #pragma unroll
      for (int mt = 0; mt < 4; ++mt) {
        const __bf16* ap = XA + (mg * 64 + mt * 16 + lr) * XA_S + k0 + hi * 8;
        v16bf af = ld_frag(ap, ap + 16);
        #pragma unroll
        for (int nt = 0; nt < 4; ++nt)
          acc[mt][nt] = __builtin_amdgcn_wmma_f32_16x16x32_bf16(
              false, af, false, bf[nt], (short)0, acc[mt][nt], false, false);
      }
    }
    // epilogue: bias + relu -> H1 (bf16 in LDS)
    #pragma unroll
    for (int mt = 0; mt < 4; ++mt) {
      #pragma unroll
      for (int nt = 0; nt < 4; ++nt) {
        const int col  = ng * 64 + nt * 16 + lr;
        const float bs = b1s[col];
        const int rb   = mg * 64 + mt * 16 + hi * 8;
        #pragma unroll
        for (int j = 0; j < 8; ++j) {
          float v = acc[mt][nt][j] + bs;
          v = v > 0.0f ? v : 0.0f;
          H1[(rb + j) * H1_S + col] = f2bf(v);
        }
      }
    }
    __syncthreads();

    // ---------------- layer 2: [128x256] @ [256x256] ----------------
    v8f acc2[4][4] = {};
    #pragma unroll
    for (int ks = 0; ks < 8; ++ks) {
      const int k0 = ks * 32;
      v16bf bf[4];
      #pragma unroll
      for (int nt = 0; nt < 4; ++nt) {
        const __bf16* bp = W2t + (ng * 64 + nt * 16 + lr) * W2_S + k0 + hi * 8;
        bf[nt] = ld_frag(bp, bp + 16);
      }
      #pragma unroll
      for (int mt = 0; mt < 4; ++mt) {
        const __bf16* ap = H1 + (mg * 64 + mt * 16 + lr) * H1_S + k0 + hi * 8;
        v16bf af = ld_frag(ap, ap + 16);
        #pragma unroll
        for (int nt = 0; nt < 4; ++nt)
          acc2[mt][nt] = __builtin_amdgcn_wmma_f32_16x16x32_bf16(
              false, af, false, bf[nt], (short)0, acc2[mt][nt], false, false);
      }
    }

    // ---------------- layer 3 fused epilogue: relu, *W3, row-reduce ----------------
    #pragma unroll
    for (int mt = 0; mt < 4; ++mt) {
      float p[8] = {0.f,0.f,0.f,0.f,0.f,0.f,0.f,0.f};
      #pragma unroll
      for (int nt = 0; nt < 4; ++nt) {
        const int col  = ng * 64 + nt * 16 + lr;
        const float bs = b2s[col];
        const float w3 = w3s[col];
        #pragma unroll
        for (int j = 0; j < 8; ++j) {
          float v = acc2[mt][nt][j] + bs;
          v = v > 0.0f ? v : 0.0f;
          p[j] += v * w3;
        }
      }
      const int rb = mg * 64 + mt * 16 + hi * 8;
      #pragma unroll
      for (int j = 0; j < 8; ++j)
        atomicAdd(&outp[rb + j], p[j]);   // ds_add_f32
    }
    __syncthreads();

    if (t < 128) {
      out[((size_t)(b0 + t)) * Dd + i] = outp[t] + b3i;
    }
    __syncthreads();   // protect outp zero/read across tile iterations
  }
}

extern "C" void kernel_launch(void* const* d_in, const int* in_sizes, int n_in,
                              void* d_out, int out_size, void* d_ws, size_t ws_size,
                              hipStream_t stream) {
  const float* inputs = (const float*)d_in[0];
  const float* adjacency = (const float*)d_in[1];
  const float* W1 = (const float*)d_in[2];
  const float* b1 = (const float*)d_in[3];
  const float* W2 = (const float*)d_in[4];
  const float* b2 = (const float*)d_in[5];
  const float* W3 = (const float*)d_in[6];
  const float* b3 = (const float*)d_in[7];
  float* out = (float*)d_out;

  const size_t shmem =
      (size_t)(Dd * XA_S + Hh * W1_S + Hh * W2_S + Dd * H1_S) * 2 /*bf16*/
      + (size_t)(3 * Hh + 128) * sizeof(float);

  (void)hipFuncSetAttribute(reinterpret_cast<const void*>(mlp_ensemble_kernel),
                            hipFuncAttributeMaxDynamicSharedMemorySize,
                            (int)shmem);

  dim3 grid(Dd, NSPLIT);
  dim3 block(256);
  mlp_ensemble_kernel<<<grid, block, shmem, stream>>>(
      inputs, adjacency, W1, b1, W2, b2, W3, b3, out);
}
